// Block_40776419508240
// MI455X (gfx1250) — matrix operand order repaired and checked by reading, not compiled, not run on hardware
//
#include <hip/hip_runtime.h>
#include <cstdint>
#include <cstddef>

// ---------------------------------------------------------------------------
// GPT-2 style transformer block for MI455X (gfx1250, wave32, WMMA + TDM).
// Compute-bound (~110 GFLOP vs ~53MB HBM traffic) -> all GEMMs and both
// attention matmuls run on v_wmma_f32_16x16x32_f16 with f32 accumulation.
// GEMM A-tiles are staged into LDS by the Tensor Data Mover (tensor_load_to_lds
// + s_wait_tensorcnt); attention is flash-style online softmax so the 800MB
// score tensor is never materialized.
// ---------------------------------------------------------------------------

#define TT 4096
#define CC 768
#define HH 12
#define HD 64
#define EPS 1e-5f

typedef _Float16 v16h __attribute__((ext_vector_type(16)));
typedef float    v8f  __attribute__((ext_vector_type(8)));
typedef unsigned int u32x4 __attribute__((ext_vector_type(4)));
typedef int          i32x4 __attribute__((ext_vector_type(4)));
typedef int          i32x8 __attribute__((ext_vector_type(8)));

#if defined(__HIP_DEVICE_COMPILE__) && __has_builtin(__builtin_amdgcn_tensor_load_to_lds)
#define USE_TDM 1
#else
#define USE_TDM 0
#endif

// Load a 16-half WMMA fragment as two 16B chunks (works for LDS or global).
static __device__ __forceinline__ v16h ld2(const _Float16* p0, const _Float16* p1) {
    union { uint4 u[2]; v16h v; } x;
    x.u[0] = *(const uint4*)p0;
    x.u[1] = *(const uint4*)p1;
    return x.v;
}

static __device__ __forceinline__ v8f wmma_f16(v16h a, v16h b, v8f c) {
    return __builtin_amdgcn_wmma_f32_16x16x32_f16(false, a, false, b, (short)0, c, false, false);
}

static __device__ __forceinline__ float gelu_exact(float v) {
    return 0.5f * v * (1.0f + erff(v * 0.70710678118654752f));
}

#if USE_TDM
// Issue a TDM DMA of a 2D tile (rows x cols, f16) global -> LDS (packed rows).
// Descriptor bit layout per CDNA5 ISA.
static __device__ __forceinline__ void tdm_load_tile_f16(
    const _Float16* gsrc, const _Float16* lds_dst,
    unsigned rows, unsigned cols, unsigned row_stride_elems) {
    const unsigned long long ga = (unsigned long long)(uintptr_t)gsrc;
    const unsigned la = (unsigned)(uintptr_t)lds_dst;   // low 32b of generic = LDS offset
    // group0: count=1 | lds_addr | global_addr[56:0] | type=2 (bits 127:126)
    u32x4 g0 = { 1u, la, (unsigned)(ga & 0xFFFFFFFFu),
                 (unsigned)((ga >> 32) & 0x01FFFFFFu) | 0x80000000u };
    // group1: data_size=1 (2B) @ bits17:16; tensor_dim0/1 huge (no OOB clip);
    // tile_dim0=cols, tile_dim1=rows, tile_dim2=0; dim0_stride=row_stride.
    const unsigned dim0 = 0x40000000u, dim1 = 0x40000000u;
    i32x8 g1;
    g1[0] = (int)(1u << 16);
    g1[1] = (int)((dim0 & 0xFFFFu) << 16);
    g1[2] = (int)(((dim0 >> 16) & 0xFFFFu) | ((dim1 & 0xFFFFu) << 16));
    g1[3] = (int)(((dim1 >> 16) & 0xFFFFu) | ((cols & 0xFFFFu) << 16));
    g1[4] = (int)(rows & 0xFFFFu);
    g1[5] = (int)row_stride_elems;          // dim0_stride[31:0] (element units)
    g1[6] = 0;                              // dim0_stride[47:32]=0 | dim1_stride lo
    g1[7] = 0;
    i32x4 z4 = { 0, 0, 0, 0 };
#if __clang_major__ >= 23
    i32x8 z8 = { 0, 0, 0, 0, 0, 0, 0, 0 };
    __builtin_amdgcn_tensor_load_to_lds(g0, g1, z4, z4, z8, 0);
#else
    __builtin_amdgcn_tensor_load_to_lds(g0, g1, z4, z4, 0);
#endif
}
#endif

// ---------------------------------------------------------------------------
// fp32 -> f16 conversion (weights)
// ---------------------------------------------------------------------------
__global__ __launch_bounds__(256) void cvt_f32_f16(const float* __restrict__ in,
                                                   _Float16* __restrict__ out, int n) {
    int i = blockIdx.x * 256 + threadIdx.x;
    if (i < n) out[i] = (_Float16)in[i];
}

// ---------------------------------------------------------------------------
// LayerNorm: one block per token row, f32 in -> f16 normalized out
// ---------------------------------------------------------------------------
__global__ __launch_bounds__(256) void layernorm_f16(const float* __restrict__ x,
                                                     const float* __restrict__ w,
                                                     const float* __restrict__ b,
                                                     _Float16* __restrict__ out) {
    const int row = blockIdx.x;
    const int tid = threadIdx.x;
    const float* xr = x + (size_t)row * CC;
    float s = 0.f, s2 = 0.f;
    for (int i = tid; i < CC; i += 256) { float v = xr[i]; s += v; s2 += v * v; }
    for (int off = 16; off > 0; off >>= 1) {
        s  += __shfl_down(s,  off, 32);
        s2 += __shfl_down(s2, off, 32);
    }
    __shared__ float rs[8], rs2[8];
    const int lane = tid & 31, wid = tid >> 5;
    if (lane == 0) { rs[wid] = s; rs2[wid] = s2; }
    __syncthreads();
    if (tid == 0) {
        float a = 0.f, a2 = 0.f;
        for (int i = 0; i < 8; ++i) { a += rs[i]; a2 += rs2[i]; }
        rs[0] = a; rs2[0] = a2;
    }
    __syncthreads();
    const float mu   = rs[0] * (1.0f / CC);
    const float var  = rs2[0] * (1.0f / CC) - mu * mu;
    const float rstd = rsqrtf(var + EPS);
    for (int i = tid; i < CC; i += 256)
        out[(size_t)row * CC + i] = (_Float16)((xr[i] - mu) * rstd * w[i] + b[i]);
}

// ---------------------------------------------------------------------------
// Tiled WMMA GEMM:  C[M,N] = A[M,K] (f16, row-major) * B[K,N] (f16, row-major)
// Block: 256 threads = 8 waves; tile 128x128; K-step 32.
// A tile (128x32) staged to LDS by the Tensor Data Mover (one DMA per step);
// B staged *transposed* [128n][32k] so both WMMA fragments are contiguous
// ds_load_b128 pairs per the CDNA5 16-bit fragment layouts.
// Wave tile: 64(M) x 32(N) -> 4x2 accumulators.
// MODE 0: out_f32 = acc + bias + residual(f32)
// MODE 1: out_f16 = gelu(acc + bias)
// MODE 2: QKV scatter: q/k key-major f16 [H][T][64], v transposed [H][64][T]
// ---------------------------------------------------------------------------
template <int MODE>
__global__ __launch_bounds__(256) void gemm_wmma(
    const _Float16* __restrict__ A, const _Float16* __restrict__ B,
    const float* __restrict__ bias, const float* __restrict__ res,
    float* __restrict__ outF, _Float16* __restrict__ outH,
    _Float16* __restrict__ qO, _Float16* __restrict__ kO, _Float16* __restrict__ vtO,
    int M, int N, int K) {
    __shared__ _Float16 lA[128 * 32];
    __shared__ _Float16 lBt[128 * 32];

    const int tid = threadIdx.x;
    const int lane = tid & 31, wid = tid >> 5;
    const int lane16 = lane & 15, hi = lane >> 4;
    const int tileM = blockIdx.y * 128, tileN = blockIdx.x * 128;
    const int wr = wid >> 2;      // 0..1 -> M offset 64*wr
    const int wc = wid & 3;       // 0..3 -> N offset 32*wc

    v8f acc[4][2];
#pragma unroll
    for (int mi = 0; mi < 4; ++mi)
#pragma unroll
        for (int ni = 0; ni < 2; ++ni)
#pragma unroll
            for (int r = 0; r < 8; ++r) acc[mi][ni][r] = 0.0f;

#if !USE_TDM
    const int arow = tid >> 1, akseg = (tid & 1) * 16;   // A: 128 rows x 2 segs
#endif
    const int brow = tid >> 3, bnseg = (tid & 7) * 16;   // B: 32 rows x 8 segs

    for (int kt = 0; kt < K; kt += 32) {
#if !USE_TDM
        const uint4* gA = (const uint4*)(A + (size_t)(tileM + arow) * K + kt + akseg);
        uint4 a0 = gA[0], a1 = gA[1];
#endif
        const uint4* gB = (const uint4*)(B + (size_t)(kt + brow) * N + tileN + bnseg);
        uint4 b0 = gB[0], b1 = gB[1];
        if (kt + 32 < K)  // hint next B tile toward the caches (global_prefetch_b8)
            __builtin_prefetch(B + (size_t)(kt + 32 + brow) * N + tileN + bnseg, 0, 1);

        __syncthreads();   // previous compute done; LDS reusable
#if USE_TDM
        if (wid == 0)      // one DMA moves the whole 128x32 A tile to LDS
            tdm_load_tile_f16(A + (size_t)tileM * K + kt, lA, 128, 32, (unsigned)K);
#else
        *(uint4*)(&lA[arow * 32 + akseg])     = a0;
        *(uint4*)(&lA[arow * 32 + akseg + 8]) = a1;
#endif
        union { uint4 u[2]; _Float16 h[16]; } bb;
        bb.u[0] = b0; bb.u[1] = b1;
#pragma unroll
        for (int j = 0; j < 16; ++j) lBt[(bnseg + j) * 32 + brow] = bb.h[j];
#if USE_TDM
        if (wid == 0) __builtin_amdgcn_s_wait_tensorcnt(0);
#endif
        __syncthreads();

        v16h af[4], bf[2];
#pragma unroll
        for (int mi = 0; mi < 4; ++mi) {
            const int m = wr * 64 + mi * 16 + lane16;
            const int k0 = hi * 8;   // A layout: lanes16-31 hold K+8 block
            af[mi] = ld2(&lA[m * 32 + k0], &lA[m * 32 + k0 + 16]);
        }
#pragma unroll
        for (int ni = 0; ni < 2; ++ni) {
            const int n = wc * 32 + ni * 16 + lane16;
            const int k0 = hi * 8;   // B layout: lanes16-31 hold K+8 block
            bf[ni] = ld2(&lBt[n * 32 + k0], &lBt[n * 32 + k0 + 16]);
        }
#pragma unroll
        for (int mi = 0; mi < 4; ++mi)
#pragma unroll
            for (int ni = 0; ni < 2; ++ni)
                acc[mi][ni] = wmma_f16(af[mi], bf[ni], acc[mi][ni]);
    }

    // Epilogue. C/D layout: elem r -> M = r + 8*hi, N = lane16.
#pragma unroll
    for (int mi = 0; mi < 4; ++mi)
#pragma unroll
        for (int ni = 0; ni < 2; ++ni)
#pragma unroll
            for (int r = 0; r < 8; ++r) {
                const int m = tileM + wr * 64 + mi * 16 + r + hi * 8;
                const int n = tileN + wc * 32 + ni * 16 + lane16;
                float v = acc[mi][ni][r] + bias[n];
                if (MODE == 0) {
                    outF[(size_t)m * N + n] = v + res[(size_t)m * N + n];
                } else if (MODE == 1) {
                    outH[(size_t)m * N + n] = (_Float16)gelu_exact(v);
                } else {  // QKV scatter
                    if (n < CC) {
                        const int hh = n >> 6, d = n & 63;
                        qO[((size_t)hh * TT + m) * HD + d] = (_Float16)v;
                    } else if (n < 2 * CC) {
                        const int n2 = n - CC, hh = n2 >> 6, d = n2 & 63;
                        kO[((size_t)hh * TT + m) * HD + d] = (_Float16)v;
                    } else {
                        const int n2 = n - 2 * CC, hh = n2 >> 6, d = n2 & 63;
                        vtO[((size_t)hh * HD + d) * TT + m] = (_Float16)v;  // V transposed
                    }
                }
            }
}

// ---------------------------------------------------------------------------
// Flash attention: grid (T/128, H), 256 threads = 8 waves.
// Each wave owns a 16-row q-tile; streams 32-key tiles with online softmax.
//   S(16x16) = Q(16x64) @ K^T  -> 2 WMMAs per key sub-tile (hd split 32+32)
//   O(16x64) += P(16x32) @ V   -> 4 WMMAs (V pre-transposed [H][64][T])
// P is re-shaped C-layout -> A-layout through per-wave LDS.
// ---------------------------------------------------------------------------
__global__ __launch_bounds__(256) void flash_attn(
    const _Float16* __restrict__ q, const _Float16* __restrict__ kk,
    const _Float16* __restrict__ vt, _Float16* __restrict__ y) {
    __shared__ _Float16 lP[8][16 * 32];

    const int tid = threadIdx.x;
    const int lane = tid & 31, wid = tid >> 5;
    const int lane16 = lane & 15, hi = lane >> 4;
    const int head  = blockIdx.y;
    const int qbase = blockIdx.x * 128 + wid * 16;

    const _Float16* qh = q  + (size_t)head * TT * HD;
    const _Float16* kh = kk + (size_t)head * TT * HD;
    const _Float16* vh = vt + (size_t)head * HD * TT;

    // Q fragments (A-layout), pre-scaled by 1/sqrt(hd) = 0.125
    v16h qa[2];
#pragma unroll
    for (int c = 0; c < 2; ++c) {
        const int m = qbase + lane16;
        const int k0 = c * 32 + hi * 8;
        qa[c] = ld2(&qh[(size_t)m * HD + k0], &qh[(size_t)m * HD + k0 + 16]);
#pragma unroll
        for (int i = 0; i < 16; ++i) qa[c][i] = qa[c][i] * (_Float16)0.125f;
    }

    float mrow[8], lrow[8];
    v8f o[4];
#pragma unroll
    for (int r = 0; r < 8; ++r) { mrow[r] = -1e30f; lrow[r] = 0.f; }
#pragma unroll
    for (int ni = 0; ni < 4; ++ni)
#pragma unroll
        for (int r = 0; r < 8; ++r) o[ni][r] = 0.0f;

    for (int kb32 = 0; kb32 <= qbase + 15; kb32 += 32) {
        // ---- S = Q @ K^T over this 32-key tile (two 16-key sub-tiles) ----
        v8f s[2];
#pragma unroll
        for (int ks = 0; ks < 2; ++ks) {
#pragma unroll
            for (int r = 0; r < 8; ++r) s[ks][r] = 0.0f;
#pragma unroll
            for (int c = 0; c < 2; ++c) {
                const int key = kb32 + ks * 16 + lane16;
                const int k0  = c * 32 + hi * 8;
                v16h bf = ld2(&kh[(size_t)key * HD + k0], &kh[(size_t)key * HD + k0 + 16]);
                s[ks] = wmma_f16(qa[c], bf, s[ks]);
            }
        }
        // ---- causal mask (only diagonal-ish tiles need it) ----
        const bool need_mask = (kb32 + 31 > qbase);
        if (need_mask) {
#pragma unroll
            for (int ks = 0; ks < 2; ++ks)
#pragma unroll
                for (int r = 0; r < 8; ++r) {
                    const int qidx = qbase + r + hi * 8;
                    const int kidx = kb32 + ks * 16 + lane16;
                    if (kidx > qidx) s[ks][r] = -1e30f;
                }
        }
        // ---- online softmax (row stats live in 16-lane groups) ----
#pragma unroll
        for (int r = 0; r < 8; ++r) {
            float mx = fmaxf(s[0][r], s[1][r]);
#pragma unroll
            for (int off = 1; off < 16; off <<= 1) mx = fmaxf(mx, __shfl_xor(mx, off, 32));
            const float mnew  = fmaxf(mrow[r], mx);
            const float alpha = __expf(mrow[r] - mnew);
            float rs = 0.f;
#pragma unroll
            for (int ks = 0; ks < 2; ++ks) {
                const float p = __expf(s[ks][r] - mnew);
                s[ks][r] = p;
                rs += p;
            }
#pragma unroll
            for (int off = 1; off < 16; off <<= 1) rs += __shfl_xor(rs, off, 32);
            lrow[r] = lrow[r] * alpha + rs;
            mrow[r] = mnew;
#pragma unroll
            for (int ni = 0; ni < 4; ++ni) o[ni][r] *= alpha;
        }
        // ---- P: C-layout -> A-layout via per-wave LDS ----
        _Float16* Pw = lP[wid];
#pragma unroll
        for (int ks = 0; ks < 2; ++ks)
#pragma unroll
            for (int r = 0; r < 8; ++r)
                Pw[(r + hi * 8) * 32 + ks * 16 + lane16] = (_Float16)s[ks][r];
        asm volatile("s_wait_dscnt 0" ::: "memory");
        v16h pa;
        {
            const int m = lane16, k0 = hi * 8;
            pa = ld2(&Pw[m * 32 + k0], &Pw[m * 32 + k0 + 16]);
        }
        // ---- O += P @ V (V transposed: contiguous over keys) ----
#pragma unroll
        for (int ni = 0; ni < 4; ++ni) {
            const int d  = ni * 16 + lane16;
            const int k0 = kb32 + hi * 8;
            v16h vf = ld2(&vh[(size_t)d * TT + k0], &vh[(size_t)d * TT + k0 + 16]);
            o[ni] = wmma_f16(pa, vf, o[ni]);
        }
        asm volatile("" ::: "memory");  // keep LDS writes of next iter behind reads
    }

    // ---- finalize: divide by row sum, store y[t][head*64+d] (f16) ----
#pragma unroll
    for (int ni = 0; ni < 4; ++ni)
#pragma unroll
        for (int r = 0; r < 8; ++r) {
            const int m = qbase + r + hi * 8;
            const int d = ni * 16 + lane16;
            y[(size_t)m * CC + head * HD + d] = (_Float16)(o[ni][r] / lrow[r]);
        }
}

// ---------------------------------------------------------------------------
// Host-side orchestration
// ---------------------------------------------------------------------------
extern "C" void kernel_launch(void* const* d_in, const int* in_sizes, int n_in,
                              void* d_out, int out_size, void* d_ws, size_t ws_size,
                              hipStream_t stream) {
    (void)in_sizes; (void)n_in; (void)out_size; (void)ws_size;
    const float* x      = (const float*)d_in[0];
    const float* ln1_w  = (const float*)d_in[1];
    const float* ln1_b  = (const float*)d_in[2];
    const float* W_attn = (const float*)d_in[3];
    const float* b_attn = (const float*)d_in[4];
    const float* W_proj = (const float*)d_in[5];
    const float* b_proj = (const float*)d_in[6];
    const float* ln2_w  = (const float*)d_in[7];
    const float* ln2_b  = (const float*)d_in[8];
    const float* W_fc   = (const float*)d_in[9];
    const float* b_fc   = (const float*)d_in[10];
    const float* W_fc2  = (const float*)d_in[11];
    const float* b_fc2  = (const float*)d_in[12];
    float* out = (float*)d_out;

    char* ws = (char*)d_ws;
    size_t off = 0;
    auto alloc = [&](size_t bytes) -> char* {
        char* p = ws + off;
        off = (off + bytes + 255) & ~(size_t)255;
        return p;
    };

    _Float16* xh   = (_Float16*)alloc((size_t)TT * CC * 2);          // ln1 out (aliased by xn2 later)
    _Float16* Wah  = (_Float16*)alloc((size_t)CC * 3 * CC * 2);
    _Float16* Wph  = (_Float16*)alloc((size_t)CC * CC * 2);
    _Float16* Wfh  = (_Float16*)alloc((size_t)CC * 4 * CC * 2);
    _Float16* Wf2h = (_Float16*)alloc((size_t)4 * CC * CC * 2);
    _Float16* qB   = (_Float16*)alloc((size_t)HH * TT * HD * 2);     // 4 x 6MB block; h aliases this
    _Float16* kB   = (_Float16*)alloc((size_t)HH * TT * HD * 2);
    _Float16* vtB  = (_Float16*)alloc((size_t)HH * HD * TT * 2);
    _Float16* yB   = (_Float16*)alloc((size_t)TT * CC * 2);
    float*    x1   = (float*)   alloc((size_t)TT * CC * 4);
    _Float16* xn2  = xh;   // xh dead after QKV GEMM
    _Float16* hB   = qB;   // q/k/v/y (4*6MB = 24MB) dead after proj; h needs T*4C*2 = 24MB

    // 1) Weight conversions to f16
    {
        int n;
        n = CC * 3 * CC;  cvt_f32_f16<<<(n + 255) / 256, 256, 0, stream>>>(W_attn, Wah, n);
        n = CC * CC;      cvt_f32_f16<<<(n + 255) / 256, 256, 0, stream>>>(W_proj, Wph, n);
        n = CC * 4 * CC;  cvt_f32_f16<<<(n + 255) / 256, 256, 0, stream>>>(W_fc,  Wfh, n);
        n = 4 * CC * CC;  cvt_f32_f16<<<(n + 255) / 256, 256, 0, stream>>>(W_fc2, Wf2h, n);
    }
    // 2) ln1
    layernorm_f16<<<TT, 256, 0, stream>>>(x, ln1_w, ln1_b, xh);
    // 3) QKV projection with scatter epilogue
    gemm_wmma<2><<<dim3(3 * CC / 128, TT / 128), 256, 0, stream>>>(
        xh, Wah, b_attn, nullptr, nullptr, nullptr, qB, kB, vtB, TT, 3 * CC, CC);
    // 4) Flash attention
    flash_attn<<<dim3(TT / 128, HH), 256, 0, stream>>>(qB, kB, vtB, yB);
    // 5) Output projection + residual -> x1 (f32)
    gemm_wmma<0><<<dim3(CC / 128, TT / 128), 256, 0, stream>>>(
        yB, Wph, b_proj, x, x1, nullptr, nullptr, nullptr, nullptr, TT, CC, CC);
    // 6) ln2
    layernorm_f16<<<TT, 256, 0, stream>>>(x1, ln2_w, ln2_b, xn2);
    // 7) FC + exact GELU -> h (f16)
    gemm_wmma<1><<<dim3(4 * CC / 128, TT / 128), 256, 0, stream>>>(
        xn2, Wfh, b_fc, nullptr, nullptr, hB, nullptr, nullptr, nullptr, TT, 4 * CC, CC);
    // 8) FC2 + residual -> out (f32)
    gemm_wmma<0><<<dim3(CC / 128, TT / 128), 256, 0, stream>>>(
        hB, Wf2h, b_fc2, x1, out, nullptr, nullptr, nullptr, nullptr, TT, CC, 4 * CC);
}
